// MultiHeadAttention_65773129171235
// MI455X (gfx1250) — matrix operand order repaired and checked by reading, not compiled, not run on hardware
//
#include <hip/hip_runtime.h>

// ---------------------------------------------------------------------------
// Causal multi-head attention for MI455X (gfx1250, wave32, WMMA f16->f32).
//   B=2, T=2048, C=1024, H=16, d=64.
// Pipeline:
//   prep:   x -> f16 ; W_qkv,W_proj -> f16 transposed [N][K]
//   qkv:    WMMA GEMM  -> Q(*1/8),K as [b,h,t,d] f16 ; V as [b,h,d,t] f16
//   attn:   flash attention, 1 wave per 16-query tile, key blocks of 32
//   proj:   WMMA GEMM  -> f32 out + bias
// ---------------------------------------------------------------------------

#define B_SZ   2
#define T_CTX  2048
#define C_EMB  1024
#define H_NUM  16
#define D_HEAD 64
#define BT     (B_SZ * T_CTX)   /* 4096 */
#define NQKV   (3 * C_EMB)      /* 3072 */

typedef _Float16 half_t;
typedef _Float16 v8h  __attribute__((ext_vector_type(8)));
typedef _Float16 v16h __attribute__((ext_vector_type(16)));
typedef float    v8f  __attribute__((ext_vector_type(8)));

__device__ __forceinline__ v8f wmma_f16(v16h a, v16h b, v8f c) {
  // D = A(16x32 f16) * B(32x16 f16) + C(16x16 f32)
  return __builtin_amdgcn_wmma_f32_16x16x32_f16(false, a, false, b, (short)0, c,
                                                false, false);
}

__device__ __forceinline__ v8f zero8() {
  v8f z = {0.f, 0.f, 0.f, 0.f, 0.f, 0.f, 0.f, 0.f};
  return z;
}

__device__ __forceinline__ v16h pack16(v8h lo, v8h hi) {
  v16h a;
#pragma unroll
  for (int i = 0; i < 8; ++i) { a[i] = lo[i]; a[8 + i] = hi[i]; }
  return a;
}

// A fragment: 16(M) x 32(K) f16.  ISA layout: lane m = lane&15,
// elems 0..7 -> K = k0..k0+7, elems 8..15 -> K = 16+k0..  with k0=(lane>>4)*8.
__device__ __forceinline__ v16h a_frag(const half_t* __restrict__ base, int ldk,
                                       int m0, int kb) {
  int lane = threadIdx.x & 31;
  int m    = lane & 15;
  int k0   = (lane >> 4) << 3;
  const half_t* p = base + (size_t)(m0 + m) * ldk + kb + k0;
  return pack16(*(const v8h*)p, *(const v8h*)(p + 16));
}

// B fragment: 32(K) x 16(N) f16 from a [N][K] buffer (B stored K-contiguous).
// ISA layout: lane n = lane&15; elems 0..7 -> K = k0..k0+7, elems 8..15 -> K = 16+k0.. with k0=(lane>>4)*8.
__device__ __forceinline__ v16h b_frag(const half_t* __restrict__ base, int ldk,
                                       int n0, int koff) {
  int lane = threadIdx.x & 31;
  int n    = lane & 15;
  int kb   = koff + ((lane >> 4) << 3);
  const half_t* p = base + (size_t)(n0 + n) * ldk + kb;
  return pack16(*(const v8h*)p, *(const v8h*)(p + 16));
}

__device__ __forceinline__ float redmax16(float v) {
  v = fmaxf(v, __shfl_xor(v, 1, 32));
  v = fmaxf(v, __shfl_xor(v, 2, 32));
  v = fmaxf(v, __shfl_xor(v, 4, 32));
  v = fmaxf(v, __shfl_xor(v, 8, 32));
  return v;
}
__device__ __forceinline__ float redsum16(float v) {
  v += __shfl_xor(v, 1, 32);
  v += __shfl_xor(v, 2, 32);
  v += __shfl_xor(v, 4, 32);
  v += __shfl_xor(v, 8, 32);
  return v;
}

// ------------------------------ prep kernels -------------------------------

__global__ __launch_bounds__(256) void convert_f16_kernel(
    const float* __restrict__ in, half_t* __restrict__ out, int n) {
  int i = blockIdx.x * blockDim.x + threadIdx.x;
  if (i < n) out[i] = (half_t)in[i];
}

// in[rows][cols] (f32) -> out[cols][rows] (f16)
__global__ __launch_bounds__(256) void transpose_f16_kernel(
    const float* __restrict__ in, half_t* __restrict__ out, int rows, int cols) {
  int i = blockIdx.x * blockDim.x + threadIdx.x;
  if (i < rows * cols) {
    int n = i / rows;
    int k = i % rows;
    out[i] = (half_t)in[(size_t)k * cols + n];
  }
}

// ------------------------------- QKV GEMM ----------------------------------
// wave tile: 64(M) x 32(N).  waves = (4096/64)*(3072/32) = 6144.

__global__ __launch_bounds__(256) void qkv_gemm_kernel(
    const half_t* __restrict__ xh, const half_t* __restrict__ wT,
    const float* __restrict__ bias, half_t* __restrict__ Q,
    half_t* __restrict__ K, half_t* __restrict__ Vt) {
  int wave  = (int)((blockIdx.x * blockDim.x + threadIdx.x) >> 5);
  int mtile = wave / (NQKV / 32);
  int ntile = wave % (NQKV / 32);
  int m0 = mtile * 64;
  int n0 = ntile * 32;
  int lane = threadIdx.x & 31;

  v8f acc[4][2];
#pragma unroll
  for (int mi = 0; mi < 4; ++mi) { acc[mi][0] = zero8(); acc[mi][1] = zero8(); }

  for (int kb = 0; kb < C_EMB; kb += 32) {
    v16h b0 = b_frag(wT, C_EMB, n0, kb);
    v16h b1 = b_frag(wT, C_EMB, n0 + 16, kb);
#pragma unroll
    for (int mi = 0; mi < 4; ++mi) {
      v16h a = a_frag(xh, C_EMB, m0 + mi * 16, kb);
      acc[mi][0] = wmma_f16(a, b0, acc[mi][0]);
      acc[mi][1] = wmma_f16(a, b1, acc[mi][1]);
    }
  }

  int rhi = (lane >> 4) << 3;
#pragma unroll
  for (int mi = 0; mi < 4; ++mi) {
#pragma unroll
    for (int ni = 0; ni < 2; ++ni) {
      int n   = n0 + ni * 16 + (lane & 15);
      int sec = n >> 10;            // 0=q 1=k 2=v
      int nn  = n & (C_EMB - 1);
      int h   = nn >> 6;
      int dd  = nn & 63;
      float bv = bias[n];
#pragma unroll
      for (int r = 0; r < 8; ++r) {
        int bt = m0 + mi * 16 + rhi + r;
        int b  = bt >> 11;
        int t  = bt & (T_CTX - 1);
        float  val = acc[mi][ni][r] + bv;
        size_t bh  = (size_t)(b * H_NUM + h);
        if (sec == 0)
          Q[(bh * T_CTX + t) * D_HEAD + dd] = (half_t)(val * 0.125f);  // fold 1/sqrt(64)
        else if (sec == 1)
          K[(bh * T_CTX + t) * D_HEAD + dd] = (half_t)val;
        else
          Vt[(bh * D_HEAD + dd) * T_CTX + t] = (half_t)val;            // V transposed
      }
    }
  }
}

// ---------------------------- flash attention ------------------------------
// 1 wave per 16-row query tile; key blocks of 32; causal (skip blocks past
// the diagonal entirely).  waves = B*H*(T/16) = 4096.

__global__ __launch_bounds__(256) void attn_kernel(
    const half_t* __restrict__ Q, const half_t* __restrict__ K,
    const half_t* __restrict__ Vt, half_t* __restrict__ O) {
  __shared__ __align__(16) half_t Pl[8][16][40];  // per-wave P tile, padded rows
  int wslot = threadIdx.x >> 5;
  int wave  = (int)((blockIdx.x * blockDim.x + threadIdx.x) >> 5);
  int bh = wave >> 7;             // 128 q-tiles per (b,h)
  int m0 = (wave & 127) << 4;
  int lane = threadIdx.x & 31;

  const half_t* Qb = Q  + (size_t)bh * T_CTX * D_HEAD;
  const half_t* Kb = K  + (size_t)bh * T_CTX * D_HEAD;
  const half_t* Vb = Vt + (size_t)bh * D_HEAD * T_CTX;

  v16h qa0 = a_frag(Qb, D_HEAD, m0, 0);
  v16h qa1 = a_frag(Qb, D_HEAD, m0, 32);

  v8f o0 = zero8(), o1 = zero8(), o2 = zero8(), o3 = zero8();
  float mrow[8], lrow[8];
#pragma unroll
  for (int r = 0; r < 8; ++r) { mrow[r] = -1e30f; lrow[r] = 0.f; }

  int rhi  = (lane >> 4) << 3;
  int kend = m0 + 16;             // causal: keys <= m0+15
  for (int kb = 0; kb < kend; kb += 32) {
    // S(16x32) = Q(16x64) @ K^T : 4 WMMAs
    v8f s1 = zero8(), s2 = zero8();
    s1 = wmma_f16(qa0, b_frag(Kb, D_HEAD, kb,      0),  s1);
    s2 = wmma_f16(qa0, b_frag(Kb, D_HEAD, kb + 16, 0),  s2);
    s1 = wmma_f16(qa1, b_frag(Kb, D_HEAD, kb,      32), s1);
    s2 = wmma_f16(qa1, b_frag(Kb, D_HEAD, kb + 16, 32), s2);

    if (kb + 32 > m0) {           // diagonal block: apply causal mask
      int c1 = kb + (lane & 15);
      int c2 = c1 + 16;
#pragma unroll
      for (int r = 0; r < 8; ++r) {
        int tq = m0 + rhi + r;
        if (c1 > tq) s1[r] = -1e30f;
        if (c2 > tq) s2[r] = -1e30f;
      }
    }

    float alpha[8];
#pragma unroll
    for (int r = 0; r < 8; ++r) {
      float v  = redmax16(fmaxf(s1[r], s2[r]));
      float mn = fmaxf(mrow[r], v);
      alpha[r] = __expf(mrow[r] - mn);
      mrow[r]  = mn;
      float p1 = __expf(s1[r] - mn);
      float p2 = __expf(s2[r] - mn);
      lrow[r]  = lrow[r] * alpha[r] + redsum16(p1 + p2);
      int mrw = rhi + r;
      Pl[wslot][mrw][lane & 15]        = (half_t)p1;   // C-layout -> LDS
      Pl[wslot][mrw][16 + (lane & 15)] = (half_t)p2;
      o0[r] *= alpha[r]; o1[r] *= alpha[r]; o2[r] *= alpha[r]; o3[r] *= alpha[r];
    }
    asm volatile("s_wait_dscnt 0" ::: "memory");       // LDS transpose fence

    v16h pa;                       // reload P in A-fragment layout
    {
      int m  = lane & 15;
      int k0 = (lane >> 4) << 3;
      const half_t* pp = &Pl[wslot][m][0];
      pa = pack16(*(const v8h*)(pp + k0), *(const v8h*)(pp + 16 + k0));
    }
    // O(16x64) += P(16x32) @ V(32x64) : 4 WMMAs, V read transposed ([d][t])
    o0 = wmma_f16(pa, b_frag(Vb, T_CTX, 0,  kb), o0);
    o1 = wmma_f16(pa, b_frag(Vb, T_CTX, 16, kb), o1);
    o2 = wmma_f16(pa, b_frag(Vb, T_CTX, 32, kb), o2);
    o3 = wmma_f16(pa, b_frag(Vb, T_CTX, 48, kb), o3);
  }

  half_t* Ob = O + (size_t)bh * T_CTX * D_HEAD;
#pragma unroll
  for (int r = 0; r < 8; ++r) {
    float inv = 1.f / lrow[r];
    int t = m0 + rhi + r;
    half_t* op = Ob + (size_t)t * D_HEAD + (lane & 15);
    op[0]  = (half_t)(o0[r] * inv);
    op[16] = (half_t)(o1[r] * inv);
    op[32] = (half_t)(o2[r] * inv);
    op[48] = (half_t)(o3[r] * inv);
  }
}

// ------------------------------- proj GEMM ---------------------------------
// A rows gather head-interleaved O[b,h,t,d]; each 32-wide k-chunk stays
// inside one head.  waves = (4096/64)*(1024/32) = 2048.

__device__ __forceinline__ v16h a_frag_heads(const half_t* __restrict__ O,
                                             int m0, int kb) {
  int lane = threadIdx.x & 31;
  int m    = lane & 15;
  int k0   = (lane >> 4) << 3;
  int bt   = m0 + m;
  int b    = bt >> 11;
  int t    = bt & (T_CTX - 1);
  int h    = kb >> 6;
  int doff = kb & 63;
  const half_t* p =
      O + ((size_t)(b * H_NUM + h) * T_CTX + t) * D_HEAD + doff + k0;
  return pack16(*(const v8h*)p, *(const v8h*)(p + 16));
}

__global__ __launch_bounds__(256) void proj_gemm_kernel(
    const half_t* __restrict__ Oh, const half_t* __restrict__ wT,
    const float* __restrict__ bias, float* __restrict__ out) {
  int wave  = (int)((blockIdx.x * blockDim.x + threadIdx.x) >> 5);
  int mtile = wave / (C_EMB / 32);
  int ntile = wave % (C_EMB / 32);
  int m0 = mtile * 64;
  int n0 = ntile * 32;
  int lane = threadIdx.x & 31;

  v8f acc[4][2];
#pragma unroll
  for (int mi = 0; mi < 4; ++mi) { acc[mi][0] = zero8(); acc[mi][1] = zero8(); }

  for (int kb = 0; kb < C_EMB; kb += 32) {
    v16h b0 = b_frag(wT, C_EMB, n0, kb);
    v16h b1 = b_frag(wT, C_EMB, n0 + 16, kb);
#pragma unroll
    for (int mi = 0; mi < 4; ++mi) {
      v16h a = a_frag_heads(Oh, m0 + mi * 16, kb);
      acc[mi][0] = wmma_f16(a, b0, acc[mi][0]);
      acc[mi][1] = wmma_f16(a, b1, acc[mi][1]);
    }
  }

  int rhi = (lane >> 4) << 3;
#pragma unroll
  for (int mi = 0; mi < 4; ++mi) {
#pragma unroll
    for (int ni = 0; ni < 2; ++ni) {
      int n = n0 + ni * 16 + (lane & 15);
      float bv = bias[n];
#pragma unroll
      for (int r = 0; r < 8; ++r) {
        int bt = m0 + mi * 16 + rhi + r;
        out[(size_t)bt * C_EMB + n] = acc[mi][ni][r] + bv;
      }
    }
  }
}

// ------------------------------- launcher ----------------------------------

extern "C" void kernel_launch(void* const* d_in, const int* in_sizes, int n_in,
                              void* d_out, int out_size, void* d_ws,
                              size_t ws_size, hipStream_t stream) {
  (void)in_sizes; (void)n_in; (void)out_size; (void)ws_size;
  const float* x      = (const float*)d_in[0];
  const float* W_qkv  = (const float*)d_in[1];
  const float* b_qkv  = (const float*)d_in[2];
  const float* W_proj = (const float*)d_in[3];
  const float* b_proj = (const float*)d_in[4];
  float* out = (float*)d_out;

  // workspace carve (~48 MB total), 256B-aligned chunks
  size_t off = 0;
  char* base = (char*)d_ws;
  auto carve = [&](size_t bytes) -> char* {
    char* p = base + off;
    off += (bytes + 255) & ~(size_t)255;
    return p;
  };
  half_t* xh     = (half_t*)carve((size_t)BT * C_EMB * 2);
  half_t* wqkvT  = (half_t*)carve((size_t)NQKV * C_EMB * 2);
  half_t* wprojT = (half_t*)carve((size_t)C_EMB * C_EMB * 2);
  half_t* Qh     = (half_t*)carve((size_t)BT * C_EMB * 2);
  half_t* Kh     = (half_t*)carve((size_t)BT * C_EMB * 2);
  half_t* Vth    = (half_t*)carve((size_t)BT * C_EMB * 2);
  half_t* Ohh    = (half_t*)carve((size_t)BT * C_EMB * 2);

  // prep
  {
    int n = BT * C_EMB;  // 4,194,304
    convert_f16_kernel<<<n / 256, 256, 0, stream>>>(x, xh, n);
  }
  {
    int n = C_EMB * NQKV;  // 3,145,728
    transpose_f16_kernel<<<n / 256, 256, 0, stream>>>(W_qkv, wqkvT, C_EMB, NQKV);
  }
  {
    int n = C_EMB * C_EMB;  // 1,048,576
    transpose_f16_kernel<<<n / 256, 256, 0, stream>>>(W_proj, wprojT, C_EMB, C_EMB);
  }

  // qkv: 6144 waves -> 768 blocks of 256 (8 waves)
  qkv_gemm_kernel<<<768, 256, 0, stream>>>(xh, wqkvT, b_qkv, Qh, Kh, Vth);
  // attention: 4096 waves -> 512 blocks
  attn_kernel<<<512, 256, 0, stream>>>(Qh, Kh, Vth, Ohh);
  // proj: 2048 waves -> 256 blocks
  proj_gemm_kernel<<<256, 256, 0, stream>>>(Ohh, wprojT, b_proj, out);
}
